// Gemma3nTextAttention_27719718928495
// MI455X (gfx1250) — matrix operand order repaired and checked by reading, not compiled, not run on hardware
//
#include <hip/hip_runtime.h>
#include <hip/hip_bf16.h>
#include <cmath>

// ---------------------------------------------------------------------------
// Gemma3n sliding-window GQA attention for MI455X (gfx1250, wave32, WMMA).
// B=2, S=2048, H=2048, NH=8, NKV=2, HD=256, WIN=512.
// All GEMMs use v_wmma_f32_16x16x32_f16. Operands are pre-materialized in f16
// in WMMA-fragment-friendly layouts (X row-major, W transposed, V transposed),
// so inner loops are pure {b128 global loads + WMMA} with no LDS staging.
// Projection kernels use 32-row x 64-col wave tiles: each B fragment feeds
// two WMMAs, halving L2 weight traffic vs 16-row tiles.
// ---------------------------------------------------------------------------

#define B_   2
#define S_   2048
#define H_   2048
#define NH_  8
#define NKV_ 2
#define HD_  256
#define WIN_ 512
#define EPS_ 1e-6f

typedef _Float16 half8 __attribute__((ext_vector_type(8)));
typedef _Float16 v16h  __attribute__((ext_vector_type(16)));
typedef float    v8f   __attribute__((ext_vector_type(8)));

static __device__ __forceinline__ v16h cat8(half8 lo, half8 hi) {
  v16h r;
#pragma unroll
  for (int i = 0; i < 8; ++i) { r[i] = lo[i]; r[i + 8] = hi[i]; }
  return r;
}

static __device__ __forceinline__ v8f wmma_f16(v16h a, v16h b, v8f c) {
  // D = A(16x32) * B(32x16) + C(16x16), f32 accum.
  return __builtin_amdgcn_wmma_f32_16x16x32_f16(
      /*neg_a=*/false, a, /*neg_b=*/false, b,
      /*c_mod=*/(short)0, c, /*reuse_a=*/false, /*reuse_b=*/false);
}

// ---------------------------------------------------------------------------
// Prep 1: X f32 -> f16 copy (vectorized b128 in / b128 out).
// ---------------------------------------------------------------------------
__global__ __launch_bounds__(256) void xcvt_kernel(
    const float* __restrict__ X, _Float16* __restrict__ Xh)
{
  size_t i = ((size_t)blockIdx.x * 256 + threadIdx.x) * 8;
  float4 a = *(const float4*)(X + i);
  float4 b = *(const float4*)(X + i + 4);
  half8 h;
  h[0] = (_Float16)a.x; h[1] = (_Float16)a.y;
  h[2] = (_Float16)a.z; h[3] = (_Float16)a.w;
  h[4] = (_Float16)b.x; h[5] = (_Float16)b.y;
  h[6] = (_Float16)b.z; h[7] = (_Float16)b.w;
  *(half8*)(Xh + i) = h;
}

// ---------------------------------------------------------------------------
// Prep 2: tiled transpose+convert  in[K][N] f32  ->  out[N][K] f16.
// grid = (N/32, K/32), block = 256. Coalesced read and write via LDS tile.
// ---------------------------------------------------------------------------
__global__ __launch_bounds__(256) void wtrans_kernel(
    const float* __restrict__ in, _Float16* __restrict__ out, int K, int N)
{
  __shared__ _Float16 tile[32][33];
  const int n0 = blockIdx.x * 32, k0 = blockIdx.y * 32;
  const int tid = threadIdx.x;
#pragma unroll
  for (int j = 0; j < 4; ++j) {
    int i = tid + 256 * j;
    int kl = i >> 5, nl = i & 31;
    tile[kl][nl] = (_Float16)in[(size_t)(k0 + kl) * N + n0 + nl];
  }
  __syncthreads();
#pragma unroll
  for (int j = 0; j < 4; ++j) {
    int i = tid + 256 * j;
    int nl = i >> 5, kl = i & 31;
    out[(size_t)(n0 + nl) * K + k0 + kl] = tile[kl][nl];
  }
}

// ---------------------------------------------------------------------------
// Kernel 1: fused QKV projection + per-head RMSNorm (+scale) + RoPE.
// grid.x = (B*S)/32 row-blocks, grid.y = 12 head-slots (8 Q, 2 K, 2 V).
// 128 thr = 4 waves; each wave owns 32 rows x 64 cols (8 WMMAs / k-chunk,
// each B fragment reused by two row halves). Direct global fragment loads.
// V is written TRANSPOSED: VT[b][kv][d][s] for direct PV B-fragments.
// ---------------------------------------------------------------------------
__global__ __launch_bounds__(128) void qkv_kernel(
    const _Float16* __restrict__ Xh, const _Float16* __restrict__ WqT,
    const _Float16* __restrict__ WkT, const _Float16* __restrict__ WvT,
    const float* __restrict__ qscale, const float* __restrict__ kscale,
    _Float16* __restrict__ Qo, _Float16* __restrict__ Ko,
    _Float16* __restrict__ VTo)
{
  __shared__ float sRED[32 * 4];   // cross-wave sum-of-squares
  __shared__ float sN[32 * 256];   // normalized f32 tile (for RoPE pairing)

  const int slot = blockIdx.y;
  const _Float16* WT; const float* scale; _Float16* outp;
  int head, nheads; bool rope;
  if (slot < 8)       { WT = WqT; head = slot;      scale = qscale; outp = Qo; nheads = NH_;  rope = true;  }
  else if (slot < 10) { WT = WkT; head = slot - 8;  scale = kscale; outp = Ko; nheads = NKV_; rope = true;  }
  else                { WT = WvT; head = slot - 10; scale = nullptr;outp = nullptr; nheads = NKV_; rope = false; }
  const size_t wrow0 = (size_t)head * HD_;  // W^T row base for this head

  const int row0 = blockIdx.x * 32;
  const int tid  = threadIdx.x;
  const int lane = tid & 31, ln = lane & 15, lh = lane >> 4;
  const int cw = tid >> 5;  // wave id == column group (0..3)

  v8f acc[2][4];
#pragma unroll
  for (int rr = 0; rr < 2; ++rr)
#pragma unroll
    for (int t = 0; t < 4; ++t)
#pragma unroll
      for (int r = 0; r < 8; ++r) acc[rr][t][r] = 0.f;

  const _Float16* arow0 = Xh + (size_t)(row0 + ln) * H_;
  const _Float16* arow1 = Xh + (size_t)(row0 + 16 + ln) * H_;
  for (int kc = 0; kc < H_; kc += 32) {
    const _Float16* a0 = arow0 + kc + lh * 16;
    const _Float16* a1 = arow1 + kc + lh * 16;
    v16h af0 = cat8(*(const half8*)a0, *(const half8*)(a0 + 8));
    v16h af1 = cat8(*(const half8*)a1, *(const half8*)(a1 + 8));
#pragma unroll
    for (int t = 0; t < 4; ++t) {
      const _Float16* bp =
          WT + (wrow0 + cw * 64 + t * 16 + ln) * (size_t)H_ + kc + lh * 16;
      v16h bf = *(const v16h*)bp;
      acc[0][t] = wmma_f16(af0, bf, acc[0][t]);
      acc[1][t] = wmma_f16(af1, bf, acc[1][t]);
    }
  }

  // ---- RMSNorm (on raw f32 accumulators, over the full 256-dim head) ----
  float ss[2][8];
#pragma unroll
  for (int rr = 0; rr < 2; ++rr)
#pragma unroll
    for (int r = 0; r < 8; ++r) {
      float s = 0.f;
#pragma unroll
      for (int t = 0; t < 4; ++t) s += acc[rr][t][r] * acc[rr][t][r];
      ss[rr][r] = s;
    }
#pragma unroll
  for (int msk = 1; msk < 16; msk <<= 1)
#pragma unroll
    for (int rr = 0; rr < 2; ++rr)
#pragma unroll
      for (int r = 0; r < 8; ++r) ss[rr][r] += __shfl_xor(ss[rr][r], msk, 32);
  if (ln == 0) {
#pragma unroll
    for (int rr = 0; rr < 2; ++rr)
#pragma unroll
      for (int r = 0; r < 8; ++r)
        sRED[(rr * 16 + r + 8 * lh) * 4 + cw] = ss[rr][r];
  }
  __syncthreads();
  float rinv[2][8];
#pragma unroll
  for (int rr = 0; rr < 2; ++rr)
#pragma unroll
    for (int r = 0; r < 8; ++r) {
      int row = rr * 16 + r + 8 * lh;
      float s = sRED[row * 4 + 0] + sRED[row * 4 + 1] +
                sRED[row * 4 + 2] + sRED[row * 4 + 3];
      rinv[rr][r] = rsqrtf(s * (1.0f / HD_) + EPS_);
    }
#pragma unroll
  for (int t = 0; t < 4; ++t) {
    int n = cw * 64 + t * 16 + ln;
    float sc = scale ? scale[n] : 1.0f;
#pragma unroll
    for (int rr = 0; rr < 2; ++rr)
#pragma unroll
      for (int r = 0; r < 8; ++r) {
        int m = rr * 16 + r + 8 * lh;
        sN[m * 256 + n] = acc[rr][t][r] * rinv[rr][r] * sc;
      }
  }
  __syncthreads();

  const int bb = row0 / S_;        // 32-row blocks never straddle batches
  const int s0 = row0 % S_;
  if (!rope) {
    // V: thread owns head-dim columns d = tid and tid+128;
    // store 32 contiguous s positions as 4x b128 each.
#pragma unroll
    for (int dd = 0; dd < 2; ++dd) {
      const int d = tid + dd * 128;
      size_t base = (((size_t)bb * NKV_ + head) * HD_ + d) * S_ + s0;
#pragma unroll
      for (int c = 0; c < 4; ++c) {
        half8 hv;
#pragma unroll
        for (int j = 0; j < 8; ++j)
          hv[j] = (_Float16)sN[(c * 8 + j) * 256 + d];
        *(half8*)(VTo + base + c * 8) = hv;
      }
    }
  } else {
    // Half-split RoPE: pair (j, j+128), inv_freq[j] = 10000^(-j/128).
    const int pj = tid;  // 128 threads == 128 pairs
    const float K1 = -logf(10000.f) * (1.0f / 128.0f);
    const float invf = expf(K1 * (float)pj);
#pragma unroll
    for (int m = 0; m < 32; ++m) {
      int spos = s0 + m;
      float fr = (float)spos * invf;
      float c = cosf(fr), s = sinf(fr);
      float x1 = sN[m * 256 + pj], x2 = sN[m * 256 + 128 + pj];
      size_t base = (((size_t)bb * nheads + head) * S_ + spos) * HD_;
      outp[base + pj]       = (_Float16)(x1 * c - x2 * s);
      outp[base + 128 + pj] = (_Float16)(x2 * c + x1 * s);
    }
  }
}

// ---------------------------------------------------------------------------
// Kernel 2: sliding-window flash attention. One wave per 16-query tile.
// grid = (S/16, NH, B), block = 32. K and V^T fragments load directly from
// global; only the P C-layout -> A-layout bounce (1KB) uses LDS.
// ---------------------------------------------------------------------------
__global__ __launch_bounds__(32) void attn_kernel(
    const _Float16* __restrict__ Q, const _Float16* __restrict__ K,
    const _Float16* __restrict__ VT, _Float16* __restrict__ O)
{
  __shared__ __attribute__((aligned(32))) _Float16 sP[16 * 32];

  const int qb = blockIdx.x, h = blockIdx.y, b = blockIdx.z;
  const int kv = h / (NH_ / NKV_);
  const int q0 = qb * 16;
  const int lane = threadIdx.x, ln = lane & 15, lh = lane >> 4;

  const _Float16* Qp  = Q  + (((size_t)b * NH_ + h) * S_ + q0) * HD_;
  const _Float16* Kp  = K  + ((size_t)b * NKV_ + kv) * S_ * HD_;
  const _Float16* VTp = VT + ((size_t)b * NKV_ + kv) * (size_t)HD_ * S_;

  // Q A-fragments for all 8 hd-chunks (kept in registers).
  v16h qa[8];
#pragma unroll
  for (int c = 0; c < 8; ++c) {
    const _Float16* p = Qp + (size_t)ln * HD_ + 32 * c + lh * 16;
    qa[c] = cat8(*(const half8*)p, *(const half8*)(p + 8));
  }

  v8f o[16];
#pragma unroll
  for (int t = 0; t < 16; ++t)
#pragma unroll
    for (int r = 0; r < 8; ++r) o[t][r] = 0.f;
  float m_r[8], l_r[8];
#pragma unroll
  for (int r = 0; r < 8; ++r) { m_r[r] = -1e30f; l_r[r] = 0.f; }

  const int klo = max(0, q0 - (WIN_ - 1)) & ~31;
  const int khi = q0 + 15;

  for (int kb = klo; kb <= khi; kb += 32) {
    // ---- scores S = Q K^T for 32 keys (two 16-wide N tiles) ----
    v8f s0, s1;
#pragma unroll
    for (int r = 0; r < 8; ++r) { s0[r] = 0.f; s1[r] = 0.f; }
    const int kr0 = min(kb + ln, S_ - 1);
    const int kr1 = min(kb + 16 + ln, S_ - 1);
#pragma unroll
    for (int c = 0; c < 8; ++c) {
      v16h b0 = *(const v16h*)(Kp + (size_t)kr0 * HD_ + 32 * c + lh * 16);
      s0 = wmma_f16(qa[c], b0, s0);
      v16h b1 = *(const v16h*)(Kp + (size_t)kr1 * HD_ + 32 * c + lh * 16);
      s1 = wmma_f16(qa[c], b1, s1);
    }
    // ---- causal + sliding-window mask (element-wise in C layout) ----
#pragma unroll
    for (int r = 0; r < 8; ++r) {
      int mabs = q0 + r + 8 * lh;
      int k0 = kb + ln, k1 = kb + 16 + ln;
      if (!(k0 <= mabs && mabs - k0 < WIN_)) s0[r] = -3.0e38f;
      if (!(k1 <= mabs && mabs - k1 < WIN_)) s1[r] = -3.0e38f;
    }
    // ---- online softmax ----
    float bm[8];
#pragma unroll
    for (int r = 0; r < 8; ++r) bm[r] = fmaxf(s0[r], s1[r]);
#pragma unroll
    for (int msk = 1; msk < 16; msk <<= 1)
#pragma unroll
      for (int r = 0; r < 8; ++r) bm[r] = fmaxf(bm[r], __shfl_xor(bm[r], msk, 32));
    float al[8];
#pragma unroll
    for (int r = 0; r < 8; ++r) {
      float mn = fmaxf(m_r[r], bm[r]);
      al[r] = __expf(m_r[r] - mn);
      m_r[r] = mn;
    }
    float rs[8];
    v8f p0, p1;
#pragma unroll
    for (int r = 0; r < 8; ++r) {
      p0[r] = __expf(s0[r] - m_r[r]);
      p1[r] = __expf(s1[r] - m_r[r]);
      rs[r] = p0[r] + p1[r];
    }
#pragma unroll
    for (int msk = 1; msk < 16; msk <<= 1)
#pragma unroll
      for (int r = 0; r < 8; ++r) rs[r] += __shfl_xor(rs[r], msk, 32);
#pragma unroll
    for (int r = 0; r < 8; ++r) l_r[r] = l_r[r] * al[r] + rs[r];
#pragma unroll
    for (int t = 0; t < 16; ++t)
#pragma unroll
      for (int r = 0; r < 8; ++r) o[t][r] *= al[r];

    // ---- P: C-layout -> LDS -> A-layout fragment ----
#pragma unroll
    for (int r = 0; r < 8; ++r) {
      int m = r + 8 * lh;
      sP[m * 32 + ln]      = (_Float16)p0[r];
      sP[m * 32 + 16 + ln] = (_Float16)p1[r];
    }
    __syncthreads();
    v16h pa = cat8(*(const half8*)&sP[ln * 32 + lh * 16],
                   *(const half8*)&sP[ln * 32 + 8 + lh * 16]);
    __syncthreads();  // P reads done before next iteration overwrites sP

    // ---- O += P V : 16 n-tiles over HD, B-frags direct from V^T ----
#pragma unroll
    for (int t = 0; t < 16; ++t) {
      v16h vb = *(const v16h*)(VTp + (size_t)(t * 16 + ln) * S_ + kb + lh * 16);
      o[t] = wmma_f16(pa, vb, o[t]);
    }
  }

  // ---- normalize and store attn output [b*S+s][h*HD+d] as f16 ----
#pragma unroll
  for (int t = 0; t < 16; ++t)
#pragma unroll
    for (int r = 0; r < 8; ++r) {
      int m = r + 8 * lh;
      float v = o[t][r] / l_r[r];
      O[((size_t)b * S_ + q0 + m) * (NH_ * HD_) + h * HD_ + t * 16 + ln] =
          (_Float16)v;
    }
}

// ---------------------------------------------------------------------------
// Kernel 3: output projection  out = O(f16) @ Wo^T(f16), f32 result.
// grid = (128 row-blocks, 8 col-blocks of 256), block = 128 (4 waves).
// Each wave: 32 rows x 64 cols, LDS-free, B frags feed two WMMAs.
// ---------------------------------------------------------------------------
__global__ __launch_bounds__(128) void oproj_kernel(
    const _Float16* __restrict__ A, const _Float16* __restrict__ WoT,
    float* __restrict__ out)
{
  const int row0 = blockIdx.x * 32;
  const int col0 = blockIdx.y * 256;
  const int tid  = threadIdx.x;
  const int lane = tid & 31, ln = lane & 15, lh = lane >> 4;
  const int cw = tid >> 5;

  v8f acc[2][4];
#pragma unroll
  for (int rr = 0; rr < 2; ++rr)
#pragma unroll
    for (int t = 0; t < 4; ++t)
#pragma unroll
      for (int r = 0; r < 8; ++r) acc[rr][t][r] = 0.f;

  const _Float16* arow0 = A + (size_t)(row0 + ln) * (NH_ * HD_);
  const _Float16* arow1 = A + (size_t)(row0 + 16 + ln) * (NH_ * HD_);
  for (int kc = 0; kc < NH_ * HD_; kc += 32) {
    const _Float16* a0 = arow0 + kc + lh * 16;
    const _Float16* a1 = arow1 + kc + lh * 16;
    v16h af0 = cat8(*(const half8*)a0, *(const half8*)(a0 + 8));
    v16h af1 = cat8(*(const half8*)a1, *(const half8*)(a1 + 8));
#pragma unroll
    for (int t = 0; t < 4; ++t) {
      const _Float16* bp =
          WoT + (size_t)(col0 + cw * 64 + t * 16 + ln) * (NH_ * HD_) + kc + lh * 16;
      v16h bf = *(const v16h*)bp;
      acc[0][t] = wmma_f16(af0, bf, acc[0][t]);
      acc[1][t] = wmma_f16(af1, bf, acc[1][t]);
    }
  }
#pragma unroll
  for (int rr = 0; rr < 2; ++rr)
#pragma unroll
    for (int t = 0; t < 4; ++t)
#pragma unroll
      for (int r = 0; r < 8; ++r) {
        int m = rr * 16 + r + 8 * lh;
        int n = cw * 64 + t * 16 + ln;
        out[(size_t)(row0 + m) * H_ + col0 + n] = acc[rr][t][r];
      }
}

// ---------------------------------------------------------------------------
// Launch. Inputs: 0 hidden_states, 1 attention_mask (== causal, applied
// analytically), 2 Wq, 3 Wk, 4 Wv, 5 Wo, 6 q_scale, 7 k_scale.
// Workspace (f16): Xh/attn-out alias (16MB) | Q 16MB | K 4MB | V^T 4MB |
// Wq^T 8MB | Wk^T 2MB | Wv^T 2MB | Wo^T 8MB  => ~63MB.
// ---------------------------------------------------------------------------
extern "C" void kernel_launch(void* const* d_in, const int* in_sizes, int n_in,
                              void* d_out, int out_size, void* d_ws,
                              size_t ws_size, hipStream_t stream) {
  (void)in_sizes; (void)n_in; (void)out_size; (void)ws_size;
  const float* X  = (const float*)d_in[0];
  const float* Wq = (const float*)d_in[2];
  const float* Wk = (const float*)d_in[3];
  const float* Wv = (const float*)d_in[4];
  const float* Wo = (const float*)d_in[5];
  const float* qs = (const float*)d_in[6];
  const float* ks = (const float*)d_in[7];

  _Float16* ws = (_Float16*)d_ws;
  size_t o = 0;
  _Float16* Xh  = ws + o; o += (size_t)B_ * S_ * H_;          // aliased as Ob
  _Float16* Qb  = ws + o; o += (size_t)B_ * NH_  * S_ * HD_;
  _Float16* Kb  = ws + o; o += (size_t)B_ * NKV_ * S_ * HD_;
  _Float16* VTb = ws + o; o += (size_t)B_ * NKV_ * S_ * HD_ + 64;  // +tail pad
  _Float16* WqT = ws + o; o += (size_t)(NH_  * HD_) * H_;
  _Float16* WkT = ws + o; o += (size_t)(NKV_ * HD_) * H_;
  _Float16* WvT = ws + o; o += (size_t)(NKV_ * HD_) * H_;
  _Float16* WoT = ws + o; o += (size_t)H_ * (NH_ * HD_);
  _Float16* Ob  = Xh;  // Xh dead after qkv_kernel; reuse for attention output

  xcvt_kernel<<<dim3((B_ * S_ * H_) / (256 * 8)), 256, 0, stream>>>(X, Xh);
  wtrans_kernel<<<dim3((NH_  * HD_) / 32, H_ / 32), 256, 0, stream>>>(
      Wq, WqT, H_, NH_ * HD_);
  wtrans_kernel<<<dim3((NKV_ * HD_) / 32, H_ / 32), 256, 0, stream>>>(
      Wk, WkT, H_, NKV_ * HD_);
  wtrans_kernel<<<dim3((NKV_ * HD_) / 32, H_ / 32), 256, 0, stream>>>(
      Wv, WvT, H_, NKV_ * HD_);
  wtrans_kernel<<<dim3(H_ / 32, (NH_ * HD_) / 32), 256, 0, stream>>>(
      Wo, WoT, NH_ * HD_, H_);

  qkv_kernel<<<dim3((B_ * S_) / 32, 12), 128, 0, stream>>>(
      Xh, WqT, WkT, WvT, qs, ks, Qb, Kb, VTb);
  attn_kernel<<<dim3(S_ / 16, NH_, B_), 32, 0, stream>>>(Qb, Kb, VTb, Ob);
  oproj_kernel<<<dim3((B_ * S_) / 32, H_ / 256), 128, 0, stream>>>(
      Ob, WoT, (float*)d_out);
}
